// MambaLayer_37108517438179
// MI455X (gfx1250) — matrix operand order repaired and checked by reading, not compiled, not run on hardware
//
#include <hip/hip_runtime.h>
#include <math.h>

// ---------------------------------------------------------------------------
// Mamba2 layer forward for MI455X (gfx1250, wave32, WMMA).
// Pipeline:
//   1) in_proj  GEMM  (bf16 WMMA 16x16x32, f32 accum)  x[16384,512] @ W_in[512,2305]
//   2) depthwise causal conv(k=4) + bias + SiLU, split into x_ssm / B / C
//   3) dt = softplus(dt_raw + bias), dA = exp(dt*A) precompute
//   4) selective scan: h in VGPRs, GLOBAL_LOAD_ASYNC_TO_LDS-staged B/C/x,
//      8 timesteps per s_wait_asynccnt+barrier
//   5) gate with silu(z) + RMSNorm (in place)
//   6) out_proj GEMM  (bf16 WMMA) -> d_out
//   7) rnn_state passthrough copy
// ---------------------------------------------------------------------------

typedef __attribute__((ext_vector_type(16))) __bf16 v16bf;
typedef __attribute__((ext_vector_type(8)))  float  v8f;

union FragU { v16bf v; uint4 q[2]; };

#define AS3(p) ((__attribute__((address_space(3))) char*)(p))
#define LDS_OFF(p) ((unsigned)(size_t)AS3(p))

// Async memory->LDS copy, tracked by ASYNCcnt (CDNA5 VGLOBAL ops, ISA ch.10/15.18).
__device__ __forceinline__ void async_copy_b64(unsigned lds_off, const void* g) {
  asm volatile("global_load_async_to_lds_b64 %0, %1, off"
               :: "v"(lds_off), "v"((unsigned long long)(size_t)g) : "memory");
}
__device__ __forceinline__ void async_copy_b128(unsigned lds_off, const void* g) {
  asm volatile("global_load_async_to_lds_b128 %0, %1, off"
               :: "v"(lds_off), "v"((unsigned long long)(size_t)g) : "memory");
}
__device__ __forceinline__ void wait_async0() {
  asm volatile("s_wait_asynccnt 0x0" ::: "memory");
}

// ---------------------------------------------------------------------------
// Generic bf16 WMMA GEMM: C[M,N] = A[M,K] (f32) * B[K,N] (f32), f32 out.
// 128x128 tile per block, 256 threads = 8 waves in 4(M) x 2(N) layout,
// each wave computes a 32x64 region = 2x4 v_wmma_f32_16x16x32_bf16 tiles.
// f32 -> bf16 conversion (v_cvt_pk_bf16_f32) happens during LDS staging
// (double buffered). Requires: M % 128 == 0, K % 32 == 0. N guarded.
// ---------------------------------------------------------------------------
__global__ __launch_bounds__(256)
void wmma_gemm_bf16(const float* __restrict__ A, const float* __restrict__ B,
                    float* __restrict__ C, int M, int N, int K,
                    int lda, int ldb, int ldc)
{
  __shared__ __attribute__((aligned(16))) __bf16 lA[2][128][32]; // [m][k]
  __shared__ __attribute__((aligned(16))) __bf16 lB[2][128][32]; // [n][k] (transposed)

  const int tid  = threadIdx.x;
  const int lane = tid & 31;
  const int wave = tid >> 5;
  const int wm   = wave >> 1;   // 0..3 -> 32-row strip
  const int wn   = wave & 1;    // 0..1 -> 64-col strip
  const int m0   = blockIdx.y * 128;
  const int n0   = blockIdx.x * 128;
  const int ksteps = K >> 5;

  v8f acc[2][4];
  #pragma unroll
  for (int mi = 0; mi < 2; ++mi)
    #pragma unroll
    for (int ni = 0; ni < 4; ++ni)
      acc[mi][ni] = (v8f){0.f,0.f,0.f,0.f,0.f,0.f,0.f,0.f};

  auto stageA = [&](int ks, int buf) {
    const int kk = ks << 5;
    #pragma unroll
    for (int i = 0; i < 4; ++i) {                 // 1024 float4 total
      int idx = tid + (i << 8);
      int r   = idx >> 3;                         // 0..127
      int c4  = (idx & 7) << 2;                   // 0,4,..,28
      const float4 v = *(const float4*)&A[(size_t)(m0 + r) * lda + kk + c4];
      lA[buf][r][c4 + 0] = (__bf16)v.x;           // packed hardware RNE cvt
      lA[buf][r][c4 + 1] = (__bf16)v.y;
      lA[buf][r][c4 + 2] = (__bf16)v.z;
      lA[buf][r][c4 + 3] = (__bf16)v.w;
    }
  };
  // Each thread owns a contiguous 16-element K half-row of one N column:
  // global reads stay coalesced across n at each k; the 16 contiguous bf16
  // LDS stores merge into 2x ds_store_b128.
  auto stageB = [&](int ks, int buf) {
    const int kk = ks << 5;
    const int n  = tid >> 1;                      // 0..127
    const int kh = (tid & 1) << 4;                // 0 or 16
    const int gn = n0 + n;
    float v[16];
    #pragma unroll
    for (int j = 0; j < 16; ++j)
      v[j] = (gn < N) ? B[(size_t)(kk + kh + j) * ldb + gn] : 0.f;
    #pragma unroll
    for (int j = 0; j < 16; ++j)
      lB[buf][n][kh + j] = (__bf16)v[j];
  };

  stageA(0, 0); stageB(0, 0);
  __syncthreads();

  for (int ks = 0; ks < ksteps; ++ks) {
    const int buf = ks & 1;
    if (ks + 1 < ksteps) { stageA(ks + 1, buf ^ 1); stageB(ks + 1, buf ^ 1); }

    // Fragment loads per CDNA5 VGPR layouts (wave32).
    // A 16x32 bf16: lane holds row M=lane&15; K elems {kbA..kbA+7, kbA+16..kbA+23}, kbA = lane<16 ? 0 : 8
    // B 32x16 bf16: lane holds col N=lane&15; K elems {kbB..kbB+7, kbB+16..kbB+23}, kbB = lane<16 ? 0 : 8
    FragU a[2], b[4];
    const int kbA = (lane < 16) ? 0 : 8;
    const int kbB = (lane < 16) ? 0 : 8;
    #pragma unroll
    for (int mi = 0; mi < 2; ++mi) {
      const int r = wm * 32 + mi * 16 + (lane & 15);
      a[mi].q[0] = *(const uint4*)&lA[buf][r][kbA];
      a[mi].q[1] = *(const uint4*)&lA[buf][r][kbA + 16];
    }
    #pragma unroll
    for (int ni = 0; ni < 4; ++ni) {
      const int c = wn * 64 + ni * 16 + (lane & 15);
      b[ni].q[0] = *(const uint4*)&lB[buf][c][kbB];
      b[ni].q[1] = *(const uint4*)&lB[buf][c][kbB + 16];
    }
    #pragma unroll
    for (int mi = 0; mi < 2; ++mi)
      #pragma unroll
      for (int ni = 0; ni < 4; ++ni)
        acc[mi][ni] = __builtin_amdgcn_wmma_f32_16x16x32_bf16(
            false, a[mi].v, false, b[ni].v, (short)0, acc[mi][ni], false, false);
    __syncthreads();
  }

  // C/D layout: VGPR r -> row (r | (lane>=16 ? 8 : 0)), col = lane&15
  const int rofs = (lane >= 16) ? 8 : 0;
  #pragma unroll
  for (int mi = 0; mi < 2; ++mi)
    #pragma unroll
    for (int ni = 0; ni < 4; ++ni) {
      const int col = n0 + wn * 64 + ni * 16 + (lane & 15);
      if (col < N) {
        #pragma unroll
        for (int r = 0; r < 8; ++r) {
          const int row = m0 + wm * 32 + mi * 16 + r + rofs;
          C[(size_t)row * ldc + col] = acc[mi][ni][r];
        }
      }
    }
}

// ---------------------------------------------------------------------------
// Depthwise causal conv1d (k=4) + bias + SiLU over xBC channels of zxbcdt,
// split into x_ssm [*,1024], B [*,128], C [*,128].
// grid = (5 channel chunks, 8 batches, 8 time segments of 256); 256 threads = channels.
// ---------------------------------------------------------------------------
__global__ __launch_bounds__(256)
void conv_silu_split(const float* __restrict__ zx, const float* __restrict__ cw,
                     const float* __restrict__ cb,
                     float* __restrict__ xo, float* __restrict__ Bo,
                     float* __restrict__ Co)
{
  const int c  = blockIdx.x * 256 + threadIdx.x;   // 0..1279
  const int b  = blockIdx.y;
  const int t0 = blockIdx.z * 256;
  const float w0 = cw[c * 4 + 0], w1 = cw[c * 4 + 1];
  const float w2 = cw[c * 4 + 2], w3 = cw[c * 4 + 3];
  const float bias = cb[c];
  const size_t rs = 2305;
  const float* src = zx + (size_t)b * 2048 * rs + 1024 + c;

  float xm3 = (t0 >= 3) ? src[(size_t)(t0 - 3) * rs] : 0.f;
  float xm2 = (t0 >= 2) ? src[(size_t)(t0 - 2) * rs] : 0.f;
  float xm1 = (t0 >= 1) ? src[(size_t)(t0 - 1) * rs] : 0.f;

  for (int tl = 0; tl < 256; ++tl) {
    const int t = t0 + tl;
    const float xc = src[(size_t)t * rs];
    float v = bias + w0 * xm3 + w1 * xm2 + w2 * xm1 + w3 * xc;
    v = v / (1.f + expf(-v));                       // SiLU
    const size_t bt = (size_t)b * 2048 + t;
    if (c < 1024)       xo[bt * 1024 + c] = v;
    else if (c < 1152)  Bo[bt * 128 + (c - 1024)] = v;
    else                Co[bt * 128 + (c - 1152)] = v;
    xm3 = xm2; xm2 = xm1; xm1 = xc;
  }
}

// ---------------------------------------------------------------------------
// dt = softplus(dt_raw + dt_bias); dA = exp(dt * A), A = -exp(A_log). NHEADS=1.
// ---------------------------------------------------------------------------
__global__ __launch_bounds__(256)
void dt_prep(const float* __restrict__ zx, const float* __restrict__ dt_bias,
             const float* __restrict__ A_log, float* __restrict__ dt2)
{
  const int bt = blockIdx.x * 256 + threadIdx.x;   // 0..16383
  const float r  = zx[(size_t)bt * 2305 + 2304] + dt_bias[0];
  const float sp = (r > 20.f) ? r : log1pf(expf(r));
  const float A  = -expf(A_log[0]);
  dt2[bt * 2 + 0] = sp;
  dt2[bt * 2 + 1] = expf(sp * A);
}

// ---------------------------------------------------------------------------
// Selective scan. grid = (16 headdim chunks of 64, 8 batches), 256 threads.
// Thread (pl = tid>>2, ng = tid&3) owns h[pl, ng*32 .. ng*32+31] in VGPRs.
// B/C/x for 8 timesteps staged with GLOBAL_LOAD_ASYNC_TO_LDS into ping-pong
// LDS buffers; s_wait_asynccnt 0 + barrier once per chunk, so the async loads
// for chunk c+1 overlap all of chunk c's serial compute.
// y (+ D*x skip) written in place over x_ssm.
// ---------------------------------------------------------------------------
__global__ __launch_bounds__(256)
void ssm_scan(float* __restrict__ xy, const float* __restrict__ Bg,
              const float* __restrict__ Cg, const float* __restrict__ dtg,
              const float* __restrict__ Dp)
{
  __shared__ __attribute__((aligned(16))) float sx[2][8][64];
  __shared__ __attribute__((aligned(16))) float sB[2][8][128];
  __shared__ __attribute__((aligned(16))) float sC[2][8][128];
  __shared__ float sdt[2][8][2];

  const int tid   = threadIdx.x;
  const int b     = blockIdx.y;
  const int pbase = blockIdx.x * 64;
  const int pl    = tid >> 2;
  const int ng    = tid & 3;
  const int nb    = ng * 32;
  const size_t bBase = (size_t)b * 2048;
  const float D0 = Dp[0];

  float h[32];
  #pragma unroll
  for (int i = 0; i < 32; ++i) h[i] = 0.f;

  // Per-thread async staging: 1x b64 (x) + 2x b128 (B, C) per chunk.
  const int stl = tid >> 5;           // timestep row 0..7 (32 threads each)
  const int sp2 = (tid & 31) << 1;    // x: 2 floats per thread (64/row)
  const int sn4 = (tid & 31) << 2;    // B/C: 4 floats per thread (128/row)

  auto stage = [&](int chunk, int buf) {
    const int t0 = chunk << 3;
    const size_t bt = bBase + t0 + stl;
    async_copy_b64 (LDS_OFF(&sx[buf][stl][sp2]), &xy[bt * 1024 + pbase + sp2]);
    async_copy_b128(LDS_OFF(&sB[buf][stl][sn4]), &Bg[bt * 128 + sn4]);
    async_copy_b128(LDS_OFF(&sC[buf][stl][sn4]), &Cg[bt * 128 + sn4]);
    if (tid < 16)
      sdt[buf][tid >> 1][tid & 1] = dtg[(bBase + t0 + (tid >> 1)) * 2 + (tid & 1)];
  };

  stage(0, 0);
  wait_async0();
  __syncthreads();

  for (int chunk = 0; chunk < 256; ++chunk) {
    const int buf = chunk & 1;
    if (chunk + 1 < 256) stage(chunk + 1, buf ^ 1);   // overlaps compute below

    #pragma unroll 1
    for (int tl = 0; tl < 8; ++tl) {
      const float dtv  = sdt[buf][tl][0];
      const float dA   = sdt[buf][tl][1];
      const float xv   = sx[buf][tl][pl];
      const float coef = dtv * xv;
      const float4* Bp = (const float4*)&sB[buf][tl][nb];
      const float4* Cp = (const float4*)&sC[buf][tl][nb];
      float accv = 0.f;
      #pragma unroll
      for (int i = 0; i < 8; ++i) {
        const float4 b4 = Bp[i], c4 = Cp[i];
        h[4*i+0] = fmaf(h[4*i+0], dA, coef * b4.x); accv = fmaf(h[4*i+0], c4.x, accv);
        h[4*i+1] = fmaf(h[4*i+1], dA, coef * b4.y); accv = fmaf(h[4*i+1], c4.y, accv);
        h[4*i+2] = fmaf(h[4*i+2], dA, coef * b4.z); accv = fmaf(h[4*i+2], c4.z, accv);
        h[4*i+3] = fmaf(h[4*i+3], dA, coef * b4.w); accv = fmaf(h[4*i+3], c4.w, accv);
      }
      accv += __shfl_xor(accv, 1, 32);             // reduce over 4 n-owning lanes
      accv += __shfl_xor(accv, 2, 32);
      if (ng == 0) {
        const int t = (chunk << 3) + tl;
        xy[(bBase + t) * 1024 + pbase + pl] = accv + D0 * xv;   // + D skip, in place
      }
    }
    wait_async0();        // chunk c+1's staging is complete
    __syncthreads();      // visible to all waves before ping-pong flips
  }
}

// ---------------------------------------------------------------------------
// y = y * silu(z); y *= rsqrt(mean(y^2) + eps) * norm_w. One block per (b,t) row.
// In place over the scan output buffer.
// ---------------------------------------------------------------------------
__global__ __launch_bounds__(256)
void gate_rmsnorm(float* __restrict__ y, const float* __restrict__ zx,
                  const float* __restrict__ nw)
{
  __shared__ float red[8];
  const int row = blockIdx.x;
  const int tid = threadIdx.x;
  const float* zrow = zx + (size_t)row * 2305;     // z = cols 0..1023
  float* yrow = y + (size_t)row * 1024;

  float v[4]; float ss = 0.f;
  #pragma unroll
  for (int i = 0; i < 4; ++i) {
    const int p = tid + (i << 8);
    const float z = zrow[p];
    const float val = yrow[p] * (z / (1.f + expf(-z)));
    v[i] = val; ss += val * val;
  }
  #pragma unroll
  for (int off = 16; off > 0; off >>= 1) ss += __shfl_xor(ss, off, 32);
  if ((tid & 31) == 0) red[tid >> 5] = ss;
  __syncthreads();
  float tot = 0.f;
  #pragma unroll
  for (int w = 0; w < 8; ++w) tot += red[w];
  const float scale = rsqrtf(tot * (1.f / 1024.f) + 1e-5f);
  #pragma unroll
  for (int i = 0; i < 4; ++i) {
    const int p = tid + (i << 8);
    yrow[p] = v[i] * scale * nw[p];
  }
}

__global__ __launch_bounds__(256)
void copy_f32(const float* __restrict__ src, float* __restrict__ dst, int n)
{
  const int i = blockIdx.x * 256 + threadIdx.x;
  if (i < n) dst[i] = src[i];
}

// ---------------------------------------------------------------------------
extern "C" void kernel_launch(void* const* d_in, const int* in_sizes, int n_in,
                              void* d_out, int out_size, void* d_ws, size_t ws_size,
                              hipStream_t stream)
{
  (void)in_sizes; (void)n_in; (void)out_size; (void)ws_size;

  const float* x       = (const float*)d_in[0];
  const float* rnn     = (const float*)d_in[1];
  const float* W_in    = (const float*)d_in[2];
  const float* conv_w  = (const float*)d_in[3];
  const float* conv_b  = (const float*)d_in[4];
  const float* dt_bias = (const float*)d_in[5];
  const float* A_log   = (const float*)d_in[6];
  const float* Dp      = (const float*)d_in[7];
  const float* norm_w  = (const float*)d_in[8];
  const float* W_out   = (const float*)d_in[9];
  float* out = (float*)d_out;

  const size_t BT = 8ull * 2048ull;                 // 16384 rows
  char* ws = (char*)d_ws;
  float* zx  = (float*)ws;                          // [16384, 2305] f32
  size_t off = BT * 2305ull * sizeof(float);
  float* xy  = (float*)(ws + off);                  // [16384, 1024] x_ssm / y (in place)
  off += BT * 1024ull * sizeof(float);
  float* Bm  = (float*)(ws + off);                  // [16384, 128]
  off += BT * 128ull * sizeof(float);
  float* Cm  = (float*)(ws + off);                  // [16384, 128]
  off += BT * 128ull * sizeof(float);
  float* dt2 = (float*)(ws + off);                  // [16384, 2] (dt, dA)

  // 1) in_proj: zx[16384,2305] = x[16384,512] @ W_in[512,2305]
  wmma_gemm_bf16<<<dim3(19, 128), 256, 0, stream>>>(x, W_in, zx,
                                                    16384, 2305, 512, 512, 2305, 2305);
  // 2) conv + SiLU + split
  conv_silu_split<<<dim3(5, 8, 8), 256, 0, stream>>>(zx, conv_w, conv_b, xy, Bm, Cm);
  // 3) dt / dA precompute
  dt_prep<<<64, 256, 0, stream>>>(zx, dt_bias, A_log, dt2);
  // 4) selective scan (writes y over x_ssm in place)
  ssm_scan<<<dim3(16, 8), 256, 0, stream>>>(xy, Bm, Cm, dt2, Dp);
  // 5) gate + RMSNorm (in place)
  gate_rmsnorm<<<16384, 256, 0, stream>>>(xy, zx, norm_w);
  // 6) out_proj: out[16384,512] = y[16384,1024] @ W_out[1024,512]
  wmma_gemm_bf16<<<dim3(4, 128), 256, 0, stream>>>(xy, W_out, out,
                                                   16384, 512, 1024, 1024, 512, 512);
  // 7) rnn_state passthrough (tuple second output)
  copy_f32<<<16, 256, 0, stream>>>(rnn, out + 8ull * 2048ull * 512ull, 8 * 512);
}
